// AttenS2S_63402307223931
// MI455X (gfx1250) — compile-verified, matrix operand order checked by reading, not run on hardware
//
#include <hip/hip_runtime.h>
#include <hip/hip_bf16.h>

#define B_   64
#define T_   20
#define H_   1280
#define EMB_ 300
#define V_   32000
#define H3_  3840          // 3*H
#define KE_  320           // EMB padded to mult of 32
#define KD_  1600          // H+EMB padded to mult of 32

typedef __attribute__((ext_vector_type(16))) __bf16 v16bf;
typedef __attribute__((ext_vector_type(8)))  float  v8f;

union FragBF { v16bf v; uint4 q[2]; };

// ---------------------------------------------------------------------------
// C[64 x N] = A[64 x Kpad](bf16) * W[N x Kpad]^T(bf16), f32 accumulate.
// 256 threads = 8 waves; wave -> (mtile = wave&3, npair = wave>>2).
// 2-way N blocking: each wave owns two 16x16 accumulators and reuses the
// A fragment across both WMMAs (halves A traffic vs 1 tile/wave).
// Workgroup covers a 64x64 output block; N must be a multiple of 64.
// Per-lane fragment addressing follows the CDNA5 16-bit A/B layout:
//   lanes 0-15 : K = k0+0..7   and k0+16..23
//   lanes 16-31: K = k0+8..15  and k0+24..31
// ---------------------------------------------------------------------------
__global__ void __launch_bounds__(256)
gemm_bf16_wmma(float* __restrict__ C,
               const __bf16* __restrict__ A,
               const __bf16* __restrict__ W,
               int N, int Kpad)
{
    const int lane = threadIdx.x & 31;
    const int wave = threadIdx.x >> 5;            // 0..7
    const int m0   = (wave & 3) << 4;             // 0,16,32,48
    const int n0   = (blockIdx.x << 6) + ((wave >> 2) << 4); // subtile 0
    const int n1   = n0 + 32;                                // subtile 1
    const int lr   = lane & 15;                   // row within 16-tile
    const int kh   = (lane >> 4) << 3;            // 0 or 8 (also C row offset)

    const __bf16* arow  = A + (size_t)(m0 + lr) * Kpad + kh;
    const __bf16* brow0 = W + (size_t)(n0 + lr) * Kpad + kh;
    const __bf16* brow1 = W + (size_t)(n1 + lr) * Kpad + kh;

    v8f acc0 = {};
    v8f acc1 = {};
#pragma unroll 2
    for (int k0 = 0; k0 < Kpad; k0 += 32) {
        // pull the streaming B rows ahead (emits global_prefetch_b8)
        __builtin_prefetch(brow0 + k0 + 128, 0, 1);
        __builtin_prefetch(brow1 + k0 + 128, 0, 1);
        FragBF fa, fb0, fb1;
        fa.q[0]  = *(const uint4*)(arow  + k0);
        fa.q[1]  = *(const uint4*)(arow  + k0 + 16);
        fb0.q[0] = *(const uint4*)(brow0 + k0);
        fb0.q[1] = *(const uint4*)(brow0 + k0 + 16);
        fb1.q[0] = *(const uint4*)(brow1 + k0);
        fb1.q[1] = *(const uint4*)(brow1 + k0 + 16);
        acc0 = __builtin_amdgcn_wmma_f32_16x16x32_bf16(
            false, fa.v, false, fb0.v, (short)0, acc0, false, false);
        acc1 = __builtin_amdgcn_wmma_f32_16x16x32_bf16(
            false, fa.v, false, fb1.v, (short)0, acc1, false, false);
    }

    float* crow0 = C + (size_t)(m0 + kh) * N + n0 + lr;
    float* crow1 = C + (size_t)(m0 + kh) * N + n1 + lr;
#pragma unroll
    for (int r = 0; r < 8; ++r) {
        crow0[(size_t)r * N] = acc0[r];
        crow1[(size_t)r * N] = acc1[r];
    }
}

// ---------------------------------------------------------------------------
// Utility fills / conversions
// ---------------------------------------------------------------------------
__global__ void fill_zero_f32(float* p, long n) {
    long i = (long)blockIdx.x * 256 + threadIdx.x;
    if (i < n) p[i] = 0.f;
}
__global__ void fill_zero_bf16(__bf16* p, long n) {
    long i = (long)blockIdx.x * 256 + threadIdx.x;
    if (i < n) p[i] = (__bf16)0.f;
}
// dst[n][0..Kpad) = bf16(src[n][0..K)) with zero pad K..Kpad
__global__ void convert_pad_bf16(const float* __restrict__ src,
                                 __bf16* __restrict__ dst,
                                 int K, int Kpad, long total)
{
    long i = (long)blockIdx.x * 256 + threadIdx.x;
    if (i >= total) return;
    long n = i / Kpad;
    int  k = (int)(i % Kpad);
    dst[i] = (k < K) ? (__bf16)src[n * K + k] : (__bf16)0.f;
}

// Encoder embedding gather -> bf16 x (B x KE_), zero-padded cols
__global__ void enc_embed_kernel(const int* __restrict__ source,
                                 const float* __restrict__ emb,
                                 __bf16* __restrict__ x_bf, int tstep)
{
    int i = blockIdx.x * 256 + threadIdx.x;      // B_*KE_ total
    if (i >= B_ * KE_) return;
    int b = i / KE_, k = i % KE_;
    int tok = source[b * T_ + tstep];
    x_bf[i] = (k < EMB_) ? (__bf16)emb[(size_t)tok * EMB_ + k] : (__bf16)0.f;
}

// ---------------------------------------------------------------------------
// GRU pointwise: h2 = (1-z)*n + z*h.  gi/gh are B x 3H (no bias yet).
// Writes f32 h (for attention / next step) and bf16 h (for WMMA GEMMs).
// Safe in-place (h_in == h_out), purely elementwise per index.
// ---------------------------------------------------------------------------
__global__ void __launch_bounds__(256)
gru_pointwise_kernel(const float* __restrict__ gi, const float* __restrict__ gh,
                     const float* __restrict__ bih, const float* __restrict__ bhh,
                     const float* h_in, float* h_out, __bf16* __restrict__ h_bf)
{
    int i = blockIdx.x * 256 + threadIdx.x;      // B_*H_ total
    if (i >= B_ * H_) return;
    int b = i / H_, j = i % H_;
    const float* gib = gi + (size_t)b * H3_;
    const float* ghb = gh + (size_t)b * H3_;
    float ir = gib[j]          + bih[j];
    float iz = gib[H_ + j]     + bih[H_ + j];
    float in = gib[2 * H_ + j] + bih[2 * H_ + j];
    float hr = ghb[j]          + bhh[j];
    float hz = ghb[H_ + j]     + bhh[H_ + j];
    float hn = ghb[2 * H_ + j] + bhh[2 * H_ + j];
    float r  = 1.f / (1.f + expf(-(ir + hr)));
    float z  = 1.f / (1.f + expf(-(iz + hz)));
    float nn = tanhf(in + r * hn);
    float h2 = (1.f - z) * nn + z * h_in[i];
    h_out[i] = h2;
    h_bf[i]  = (__bf16)h2;
}

// ---------------------------------------------------------------------------
// Attention pieces (replicating the reference's reshape/axis semantics)
// ---------------------------------------------------------------------------
// enc_we[t*B+b] = dot(enc_flat[(t*B+b)*H ..], wa_e)   (step-invariant)
__global__ void __launch_bounds__(256)
enc_we_kernel(const float* __restrict__ enc_flat,
              const float* __restrict__ attn_W, float* __restrict__ enc_we)
{
    int f = blockIdx.x;                           // 0..T_*B_
    int tid = threadIdx.x;
    const float* row  = enc_flat + (size_t)f * H_;
    const float* wa_e = attn_W + H_;
    __shared__ float s[256];
    float p = 0.f;
    for (int j = tid; j < H_; j += 256) p += row[j] * wa_e[j];
    s[tid] = p; __syncthreads();
    for (int st = 128; st > 0; st >>= 1) {
        if (tid < st) s[tid] += s[tid + st];
        __syncthreads();
    }
    if (tid == 0) enc_we[f] = s[0];
}

// hd[b] = dot(h[b], wa_h)
__global__ void __launch_bounds__(256)
hdot_kernel(const float* __restrict__ h, const float* __restrict__ attn_W,
            float* __restrict__ hd)
{
    int b = blockIdx.x, tid = threadIdx.x;
    __shared__ float s[256];
    float p = 0.f;
    for (int j = tid; j < H_; j += 256) p += h[(size_t)b * H_ + j] * attn_W[j];
    s[tid] = p; __syncthreads();
    for (int st = 128; st > 0; st >>= 1) {
        if (tid < st) s[tid] += s[tid + st];
        __syncthreads();
    }
    if (tid == 0) hd[b] = s[0];
}

// Reference does softmax over axis=1 of scores(T,B): softmax over the BATCH
// axis per t.  aw[t*B+b] = softmax_b(enc_we[t*B+b] + hd[b] + attn_b).
__global__ void __launch_bounds__(64)
softmax_b_kernel(const float* __restrict__ enc_we, const float* __restrict__ hd,
                 const float* __restrict__ attn_b, float* __restrict__ aw)
{
    int t = blockIdx.x, b = threadIdx.x;          // 64 threads
    __shared__ float s[64];
    float v = enc_we[t * B_ + b] + hd[b] + attn_b[0];
    s[b] = v; __syncthreads();
    for (int st = 32; st > 0; st >>= 1) {
        if (b < st) s[b] = fmaxf(s[b], s[b + st]);
        __syncthreads();
    }
    float mx = s[0]; __syncthreads();
    float e = expf(v - mx);
    s[b] = e; __syncthreads();
    for (int st = 32; st > 0; st >>= 1) {
        if (b < st) s[b] += s[b + st];
        __syncthreads();
    }
    aw[t * B_ + b] = e / s[0];
}

// Reference reshapes (T,B)->(B,T) and (T,B,H)->(B,T,H) flat (NOT transpose):
//   ctx[b][j] = sum_t aw_flat[b*T+t] * enc_flat[(b*T+t)*H + j]
// Builds x = relu(concat(ctx, emb)) as bf16 padded to KD_, writes ws output.
__global__ void __launch_bounds__(256)
ctx_build_kernel(const float* __restrict__ aw, const float* __restrict__ enc_flat,
                 const float* __restrict__ dec_emb, const int* __restrict__ target,
                 int tstep, __bf16* __restrict__ x_bf, float* __restrict__ ws_out)
{
    int b = blockIdx.x, tid = threadIdx.x;
    __shared__ float a[T_];
    if (tid < T_) {
        float av = aw[b * T_ + tid];
        a[tid] = av;
        ws_out[(size_t)tstep * B_ * T_ + b * T_ + tid] = av;
    }
    __syncthreads();
    for (int j = tid; j < H_; j += 256) {
        float c = 0.f;
#pragma unroll
        for (int t = 0; t < T_; ++t)
            c += a[t] * enc_flat[((size_t)b * T_ + t) * H_ + j];
        x_bf[(size_t)b * KD_ + j] = (__bf16)fmaxf(c, 0.f);
    }
    int tok = target[b * T_ + (tstep ? tstep - 1 : 0)];
    for (int j = tid; j < EMB_; j += 256)
        x_bf[(size_t)b * KD_ + H_ + j] =
            (__bf16)fmaxf(dec_emb[(size_t)tok * EMB_ + j], 0.f);
    for (int j = H_ + EMB_ + tid; j < KD_; j += 256)
        x_bf[(size_t)b * KD_ + j] = (__bf16)0.f;
}

// out[b][tstep][v] = (logits[b][v]+bias[v]) - logsumexp_v(...)
__global__ void __launch_bounds__(256)
logsoftmax_kernel(const float* __restrict__ logits, const float* __restrict__ bias,
                  float* __restrict__ out, int tstep)
{
    int b = blockIdx.x, tid = threadIdx.x;
    __shared__ float s[256];
    const float* lrow = logits + (size_t)b * V_;
    float mx = -3.4e38f;
    for (int v = tid; v < V_; v += 256) mx = fmaxf(mx, lrow[v] + bias[v]);
    s[tid] = mx; __syncthreads();
    for (int st = 128; st > 0; st >>= 1) {
        if (tid < st) s[tid] = fmaxf(s[tid], s[tid + st]);
        __syncthreads();
    }
    mx = s[0]; __syncthreads();
    float sum = 0.f;
    for (int v = tid; v < V_; v += 256) sum += expf(lrow[v] + bias[v] - mx);
    s[tid] = sum; __syncthreads();
    for (int st = 128; st > 0; st >>= 1) {
        if (tid < st) s[tid] += s[tid + st];
        __syncthreads();
    }
    float lse = logf(s[0]) + mx;
    float* orow = out + ((size_t)b * T_ + tstep) * V_;
    for (int v = tid; v < V_; v += 256) orow[v] = lrow[v] + bias[v] - lse;
}

// ---------------------------------------------------------------------------
// Host: orchestrate the recurrent schedule with stream-ordered launches.
// ---------------------------------------------------------------------------
extern "C" void kernel_launch(void* const* d_in, const int* in_sizes, int n_in,
                              void* d_out, int out_size, void* d_ws, size_t ws_size,
                              hipStream_t stream)
{
    const int*   source  = (const int*)  d_in[0];
    const int*   target  = (const int*)  d_in[1];
    const float* enc_emb = (const float*)d_in[2];
    const float* enc_Wih = (const float*)d_in[3];
    const float* enc_Whh = (const float*)d_in[4];
    const float* enc_bih = (const float*)d_in[5];
    const float* enc_bhh = (const float*)d_in[6];
    const float* dec_emb = (const float*)d_in[7];
    const float* attn_W  = (const float*)d_in[8];
    const float* attn_b  = (const float*)d_in[9];
    const float* dec_Wih = (const float*)d_in[10];
    const float* dec_Whh = (const float*)d_in[11];
    const float* dec_bih = (const float*)d_in[12];
    const float* dec_bhh = (const float*)d_in[13];
    const float* dense_W = (const float*)d_in[14];
    const float* dense_b = (const float*)d_in[15];

    float* out_dec = (float*)d_out;                       // (B,T,V)
    float* out_ws  = out_dec + (size_t)B_ * T_ * V_;      // (T,B,T)

    // ---- workspace carve-out (256B aligned slabs) ----
    char* wptr = (char*)d_ws;
    auto alloc = [&](size_t bytes) -> char* {
        char* p = wptr;
        wptr += (bytes + 255) & ~(size_t)255;
        return p;
    };
    __bf16* wEih  = (__bf16*)alloc((size_t)H3_ * KE_ * 2);   // enc_Wih  bf16, K padded
    __bf16* wEhh  = (__bf16*)alloc((size_t)H3_ * H_  * 2);   // enc_Whh  bf16
    __bf16* wDih  = (__bf16*)alloc((size_t)H3_ * KD_ * 2);   // dec_Wih  bf16, K padded
    __bf16* wDhh  = (__bf16*)alloc((size_t)H3_ * H_  * 2);   // dec_Whh  bf16
    __bf16* wV    = (__bf16*)alloc((size_t)V_  * H_  * 2);   // dense_W  bf16 (82MB; L2-resident)
    __bf16* x_bf  = (__bf16*)alloc((size_t)B_ * KD_ * 2);    // GEMM A operand (enc uses KE_ stride)
    __bf16* h_bf  = (__bf16*)alloc((size_t)B_ * H_ * 2);     // bf16 hidden state
    float*  gi    = (float*) alloc((size_t)B_ * H3_ * 4);
    float*  gh    = (float*) alloc((size_t)B_ * H3_ * 4);
    float*  encO  = (float*) alloc((size_t)T_ * B_ * H_ * 4);// scan outputs (T,B,H) flat
    float*  encWe = (float*) alloc((size_t)T_ * B_ * 4);     // enc @ wa_e   (T,B) flat
    float*  h_dec = (float*) alloc((size_t)B_ * H_ * 4);
    float*  h_zero= (float*) alloc((size_t)B_ * H_ * 4);
    float*  hd    = (float*) alloc((size_t)B_ * 4);
    float*  aw    = (float*) alloc((size_t)T_ * B_ * 4);
    float*  logits= (float*) alloc((size_t)B_ * V_ * 4);

    auto blks = [](long n) { return (unsigned)((n + 255) / 256); };

    // ---- one-time (per call) weight conversion to bf16, K zero-padded ----
    { long t = (long)H3_ * KE_; convert_pad_bf16<<<blks(t), 256, 0, stream>>>(enc_Wih, wEih, EMB_,      KE_, t); }
    { long t = (long)H3_ * H_;  convert_pad_bf16<<<blks(t), 256, 0, stream>>>(enc_Whh, wEhh, H_,        H_,  t); }
    { long t = (long)H3_ * KD_; convert_pad_bf16<<<blks(t), 256, 0, stream>>>(dec_Wih, wDih, H_ + EMB_, KD_, t); }
    { long t = (long)H3_ * H_;  convert_pad_bf16<<<blks(t), 256, 0, stream>>>(dec_Whh, wDhh, H_,        H_,  t); }
    { long t = (long)V_  * H_;  convert_pad_bf16<<<blks(t), 256, 0, stream>>>(dense_W, wV,   H_,        H_,  t); }

    fill_zero_f32 <<<blks((long)B_ * H_), 256, 0, stream>>>(h_zero, (long)B_ * H_);
    fill_zero_bf16<<<blks((long)B_ * H_), 256, 0, stream>>>(h_bf,   (long)B_ * H_);

    // ================= Encoder: 20 sequential GRU steps =================
    const float* h_prev = h_zero;
    for (int t = 0; t < T_; ++t) {
        float* h_next = encO + (size_t)t * B_ * H_;
        enc_embed_kernel<<<blks((long)B_ * KE_), 256, 0, stream>>>(source, enc_emb, x_bf, t);
        gemm_bf16_wmma<<<H3_ / 64, 256, 0, stream>>>(gi, x_bf, wEih, H3_, KE_);
        gemm_bf16_wmma<<<H3_ / 64, 256, 0, stream>>>(gh, h_bf, wEhh, H3_, H_);
        gru_pointwise_kernel<<<blks((long)B_ * H_), 256, 0, stream>>>(
            gi, gh, enc_bih, enc_bhh, h_prev, h_next, h_bf);
        h_prev = h_next;
    }

    // Step-invariant attention key dots: enc_we[t*B+b]
    enc_we_kernel<<<T_ * B_, 256, 0, stream>>>(encO, attn_W, encWe);

    // ================= Decoder: 20 sequential steps =================
    fill_zero_f32 <<<blks((long)B_ * H_), 256, 0, stream>>>(h_dec, (long)B_ * H_);
    fill_zero_bf16<<<blks((long)B_ * H_), 256, 0, stream>>>(h_bf,  (long)B_ * H_);

    for (int t = 0; t < T_; ++t) {
        hdot_kernel     <<<B_, 256, 0, stream>>>(h_dec, attn_W, hd);
        softmax_b_kernel<<<T_,  64, 0, stream>>>(encWe, hd, attn_b, aw);
        ctx_build_kernel<<<B_, 256, 0, stream>>>(aw, encO, dec_emb, target, t, x_bf, out_ws);
        gemm_bf16_wmma<<<H3_ / 64, 256, 0, stream>>>(gi, x_bf, wDih, H3_, KD_);
        gemm_bf16_wmma<<<H3_ / 64, 256, 0, stream>>>(gh, h_bf, wDhh, H3_, H_);
        gru_pointwise_kernel<<<blks((long)B_ * H_), 256, 0, stream>>>(
            gi, gh, dec_bih, dec_bhh, h_dec, h_dec, h_bf);
        gemm_bf16_wmma<<<V_ / 64, 256, 0, stream>>>(logits, h_bf, wV, V_, H_);
        logsoftmax_kernel<<<B_, 256, 0, stream>>>(logits, dense_b, out_dec, t);
    }
}
